// CGCNNConv_6270652252664
// MI455X (gfx1250) — compile-verified, matrix operand order checked by reading, not run on hardware
//
#include <hip/hip_runtime.h>
#include <hip/hip_bf16.h>
#include <math.h>

#define N_NODES   50000
#define N_EDGES   800000
#define NODE_DIM  128
#define EDGE_DIM  10
#define IN_DIM    266          // 2*128 + 10
#define K_PAD     288          // padded to 9 * 32
#define KSTEPS    9            // 288 / 32
#define NTILES    8            // 128 / 16
#define FRAG_PER_MAT (NTILES * KSTEPS * 32 * 16)   // 36864 bf16 per weight matrix
#define OUT_ELEMS (N_NODES * NODE_DIM)             // 6,400,000

typedef __attribute__((ext_vector_type(16))) __bf16 v16bf;
typedef __attribute__((ext_vector_type(8)))  __bf16 v8bf;
typedef __attribute__((ext_vector_type(8)))  float  v8f;

__device__ __forceinline__ unsigned short f2bf(float f) {
    unsigned int u = __float_as_uint(f);
    unsigned int r = u + 0x7FFFu + ((u >> 16) & 1u);   // round-to-nearest-even
    return (unsigned short)(r >> 16);
}

__device__ __forceinline__ float softplus_f(float x) {
    return fmaxf(x, 0.f) + log1pf(__expf(-fabsf(x)));
}

// ---------------------------------------------------------------------------
// 1) Repack fp32 weights [266 x 128] -> bf16 WMMA B-fragment layout, K padded
//    to 288.  Fragment element (t, s, lane, j) maps to (n, k):
//      n = t*16 + lane%16
//      lanes 0-15 : j<8 -> k = 32s + j        ; j>=8 -> k = 32s + 16 + (j-8)
//      lanes 16-31: j<8 -> k = 32s + 8 + j    ; j>=8 -> k = 32s + 24 + (j-8)
//    (matches the CDNA5 16-bit A/B VGPR striping).
// ---------------------------------------------------------------------------
__global__ void prep_weights(const float* __restrict__ gate_w,
                             const float* __restrict__ cand_w,
                             unsigned short* __restrict__ frag) {
    int tid = blockIdx.x * blockDim.x + threadIdx.x;
    if (tid >= 2 * FRAG_PER_MAT) return;
    int mat  = tid / FRAG_PER_MAT;
    int r    = tid % FRAG_PER_MAT;
    int t    = r / (KSTEPS * 512);
    int r2   = r % (KSTEPS * 512);
    int s    = r2 / 512;
    int l2   = r2 % 512;
    int lane = l2 / 16;
    int j    = l2 % 16;
    int n    = t * 16 + (lane & 15);
    int k0   = s * 32;
    int half = (lane < 16) ? 0 : 8;
    int k    = (j < 8) ? (k0 + half + j) : (k0 + 16 + half + (j - 8));
    const float* w = mat ? cand_w : gate_w;
    float v = (k < IN_DIM) ? w[k * NODE_DIM + n] : 0.f;
    frag[tid] = f2bf(v);
}

__global__ void zero_f(float* __restrict__ p, int n) {
    int i = blockIdx.x * blockDim.x + threadIdx.x;
    if (i < n) p[i] = 0.f;
}

// ---------------------------------------------------------------------------
// 2) Main edge kernel: one block per 16-edge tile; 8 waves x 16 output cols.
// ---------------------------------------------------------------------------
__global__ __launch_bounds__(256) void edge_kernel(
        const float* __restrict__ h,
        const int* __restrict__ eidx,            // [2 x N_EDGES]
        const float* __restrict__ ef,            // [N_EDGES x 10]
        const unsigned short* __restrict__ frag, // bf16 fragments (gate|cand)
        const float* __restrict__ gate_b,
        const float* __restrict__ cand_b,
        float* __restrict__ agg) {
    __shared__ unsigned short zt[16][296];       // row stride padded: 592B/row
    __shared__ int s_src[16];
    __shared__ int s_dst[16];

    const int tile = blockIdx.x;
    const int e0   = tile * 16;
    const int t    = threadIdx.x;

    if (t < 16) {
        int v = eidx[e0 + t];
        if ((unsigned)v >= (unsigned)N_NODES) v = 0;
        s_src[t] = v;
    } else if (t < 32) {
        int v = eidx[N_EDGES + e0 + (t - 16)];
        if ((unsigned)v >= (unsigned)N_NODES) v = 0;
        s_dst[t - 16] = v;
    }
    __syncthreads();

    // Stage z tile: [h[src] | h[dst] | edge_feat | zero-pad] as bf16.
    for (int i = t; i < 16 * K_PAD; i += 256) {
        int row = i / K_PAD;
        int col = i - row * K_PAD;
        float v;
        if (col < NODE_DIM)            v = h[s_src[row] * NODE_DIM + col];
        else if (col < 2 * NODE_DIM)   v = h[s_dst[row] * NODE_DIM + (col - NODE_DIM)];
        else if (col < IN_DIM)         v = ef[(e0 + row) * EDGE_DIM + (col - 2 * NODE_DIM)];
        else                           v = 0.f;
        zt[row][col] = f2bf(v);
    }
    __syncthreads();

    const int w     = t >> 5;        // wave id -> n-tile
    const int lane  = t & 31;
    const int nloc  = lane & 15;
    const int n     = w * 16 + nloc;
    const int mhalf = (lane >= 16) ? 8 : 0;
    const int arow  = lane & 15;     // A-fragment M row for this lane
    const int ahalf = (lane < 16) ? 0 : 8;

    const unsigned short* gfrag = frag + (size_t)w * (KSTEPS * 512);
    const unsigned short* cfrag = frag + FRAG_PER_MAT + (size_t)w * (KSTEPS * 512);

    v8f accg = {};
    v8f accc = {};

    #pragma unroll
    for (int s = 0; s < KSTEPS; ++s) {
        const int k0 = s * 32;
        const __bf16* rp = reinterpret_cast<const __bf16*>(&zt[arow][0]);
        v8bf lo = *reinterpret_cast<const v8bf*>(rp + k0 + ahalf);
        v8bf hi = *reinterpret_cast<const v8bf*>(rp + k0 + 16 + ahalf);
        v16bf a = __builtin_shufflevector(lo, hi,
                    0, 1, 2, 3, 4, 5, 6, 7, 8, 9, 10, 11, 12, 13, 14, 15);
        v16bf bg = *reinterpret_cast<const v16bf*>(gfrag + (s * 512 + lane * 16));
        v16bf bc = *reinterpret_cast<const v16bf*>(cfrag + (s * 512 + lane * 16));
        accg = __builtin_amdgcn_wmma_f32_16x16x32_bf16(false, a, false, bg,
                                                       (short)0, accg, false, false);
        accc = __builtin_amdgcn_wmma_f32_16x16x32_bf16(false, a, false, bc,
                                                       (short)0, accc, false, false);
    }

    const float gb = gate_b[n];
    const float cb = cand_b[n];
    #pragma unroll
    for (int r = 0; r < 8; ++r) {
        const int m = r + mhalf;                       // edge row within tile
        float g  = 1.f / (1.f + __expf(-(accg[r] + gb)));
        float cv = softplus_f(accc[r] + cb);
        atomicAdd(&agg[(size_t)s_src[m] * NODE_DIM + n], g * cv);
    }
}

// ---------------------------------------------------------------------------
// 3) BatchNorm statistics: one block per channel.
// ---------------------------------------------------------------------------
__global__ __launch_bounds__(256) void bn_stats(const float* __restrict__ agg,
                                                float* __restrict__ stats) {
    const int c = blockIdx.x;
    __shared__ float ssum[256];
    __shared__ float ssq[256];
    float s = 0.f, q = 0.f;
    for (int r = threadIdx.x; r < N_NODES; r += 256) {
        float v = agg[(size_t)r * NODE_DIM + c];
        s += v;
        q += v * v;
    }
    ssum[threadIdx.x] = s;
    ssq[threadIdx.x]  = q;
    __syncthreads();
    for (int o = 128; o > 0; o >>= 1) {
        if (threadIdx.x < o) {
            ssum[threadIdx.x] += ssum[threadIdx.x + o];
            ssq[threadIdx.x]  += ssq[threadIdx.x + o];
        }
        __syncthreads();
    }
    if (threadIdx.x == 0) {
        float mean = ssum[0] / (float)N_NODES;
        float var  = ssq[0] / (float)N_NODES - mean * mean;
        stats[c]             = mean;
        stats[NODE_DIM + c]  = rsqrtf(fmaxf(var, 0.f) + 1e-5f);
    }
}

// ---------------------------------------------------------------------------
// 4) out = softplus(h + bn(agg))  (in place on d_out which holds agg)
// ---------------------------------------------------------------------------
__global__ void bn_act(const float* __restrict__ h,
                       const float* __restrict__ stats,
                       const float* __restrict__ gamma,
                       const float* __restrict__ beta,
                       float* __restrict__ out) {
    int i = blockIdx.x * blockDim.x + threadIdx.x;
    if (i >= OUT_ELEMS) return;
    int c = i & (NODE_DIM - 1);
    float a  = out[i];
    float bn = (a - stats[c]) * stats[NODE_DIM + c] * gamma[c] + beta[c];
    out[i] = softplus_f(h[i] + bn);
}

extern "C" void kernel_launch(void* const* d_in, const int* in_sizes, int n_in,
                              void* d_out, int out_size, void* d_ws, size_t ws_size,
                              hipStream_t stream) {
    const float* h       = (const float*)d_in[0];
    const int*   eidx    = (const int*)  d_in[1];
    const float* ef      = (const float*)d_in[2];
    const float* gate_w  = (const float*)d_in[3];
    const float* gate_b  = (const float*)d_in[4];
    const float* cand_w  = (const float*)d_in[5];
    const float* cand_b  = (const float*)d_in[6];
    const float* gamma   = (const float*)d_in[7];
    const float* beta    = (const float*)d_in[8];
    float* out = (float*)d_out;

    unsigned short* frag = (unsigned short*)d_ws;                // 2*36864 bf16
    float* stats = (float*)((char*)d_ws + 2 * FRAG_PER_MAT * sizeof(unsigned short));

    // 1) repack weights to bf16 fragments
    prep_weights<<<(2 * FRAG_PER_MAT + 255) / 256, 256, 0, stream>>>(gate_w, cand_w, frag);
    // 2) zero the aggregation buffer (d_out doubles as agg accumulator)
    zero_f<<<(OUT_ELEMS + 255) / 256, 256, 0, stream>>>(out, OUT_ELEMS);
    // 3) gather + WMMA gate/cand + gated message + atomic scatter-add
    edge_kernel<<<N_EDGES / 16, 256, 0, stream>>>(h, eidx, ef, frag, gate_b, cand_b, out);
    // 4) batchnorm statistics
    bn_stats<<<NODE_DIM, 256, 0, stream>>>(out, stats);
    // 5) normalize + residual + softplus
    bn_act<<<(OUT_ELEMS + 255) / 256, 256, 0, stream>>>(h, stats, gamma, beta, out);
}
